// ClusterNet_51582557225284
// MI455X (gfx1250) — matrix operand order repaired and checked by reading, not compiled, not run on hardware
//
#include <hip/hip_runtime.h>

// ---------------------------------------------------------------------------
// ClusterNet forward for MI455X (gfx1250, wave32, WMMA + TDM).
//  - GEMM1/2/3: v_wmma_f32_16x16x32_bf16, f32 accumulate.
//  - A tiles staged into LDS by the Tensor Data Mover (double buffered,
//    TDM applies the bank-conflict padding via D# pad fields).
//  - B matrices pre-packed into fragment layout; frags are two
//    global_load_b128 from L2 (weights are <4 MB total).
//  - Sinkhorn as diag(u)*M*diag(v) matvecs; M = exp(-dist/eps) stored once
//    (128 MB < 192 MB L2 -> iterations run at L2 bandwidth).
// ---------------------------------------------------------------------------

typedef __bf16 bf16;
typedef __attribute__((ext_vector_type(16))) __bf16 v16bf;
typedef __attribute__((ext_vector_type(8)))  __bf16 v8bf;
typedef __attribute__((ext_vector_type(8)))  float  v8f;
typedef __attribute__((ext_vector_type(4)))  unsigned int v4u;
typedef __attribute__((ext_vector_type(8)))  int v8i;
typedef __attribute__((ext_vector_type(4)))  int v4i;

#define Bsz   32768
#define DIN   768
#define Hdim  2048
#define NZ    128
#define NCL   1024
#define EPSI  0.5f
#define NIT   15

#define BLK_M 128
#define BLK_K 32
#define LDA_S 40    // padded LDS stride (halves): 32 data + 4-dword pad

#if defined(__has_builtin)
#  if __has_builtin(__builtin_amdgcn_tensor_load_to_lds)
#    define HAVE_TDM 1
#  endif
#endif
#ifndef HAVE_TDM
#  define HAVE_TDM 0
#endif

// ------------------------- WMMA fragment loaders ---------------------------
__device__ inline v16bf cat8(v8bf lo, v8bf hi) {
  return __builtin_shufflevector(lo, hi, 0, 1, 2, 3, 4, 5, 6, 7,
                                 8, 9, 10, 11, 12, 13, 14, 15);
}

// 16-bit A 16x32 (MxK): lane<16 row=lane holds K 0..7,16..23; lane>=16 holds
// K 8..15,24..31 (ISA 7.12.2). Tile in LDS row-major, stride LDA_S halves.
__device__ inline v16bf load_a_frag(const bf16* s) {
  int lane = threadIdx.x & 31;
  const bf16* p = s + (lane & 15) * LDA_S + (lane >> 4) * 8;
  return cat8(*(const v8bf*)p, *(const v8bf*)(p + 16));
}

// B tile pre-packed as [tile][n(16)][k(32)]; each lane (n=lane&15) takes
// k = (lane>>4)*8 ..+7 and 16+(lane>>4)*8 ..+7: two b128 loads.
__device__ inline v16bf load_b_frag_packed(const bf16* pk) {
  int lane = threadIdx.x & 31;
  const bf16* p = pk + (lane & 15) * 32 + (lane >> 4) * 8;
  return cat8(*(const v8bf*)p, *(const v8bf*)(p + 16));
}

__device__ inline unsigned lds_off(const void* p) {
  return (unsigned)(unsigned long long)p;   // low 32 bits = LDS byte address
}

// ------------------------- TDM tile load (A: 128 x 32 bf16) ----------------
__device__ inline void tdm_load_tileA(const bf16* gsrc, unsigned ldsAddr, int lda) {
#if HAVE_TDM
  unsigned long long ga = (unsigned long long)gsrc;
  v4u g0;
  g0[0] = 1u;                                   // count=1, user mode
  g0[1] = ldsAddr;                              // lds_addr
  g0[2] = (unsigned)ga;                         // global_addr[31:0]
  g0[3] = (unsigned)((ga >> 32) & 0x1FFFFFFu) | (2u << 30);  // addr hi | type=2
  v8i g1;
  // data_size=2B; pad_enable, 4 dwords pad every 16 dwords (-> LDA_S=40)
  g1[0] = (int)((1u << 16) | (1u << 20) | (3u << 22) | (3u << 25));
  g1[1] = (int)(32u << 16);                     // tensor_dim0 = 32
  g1[2] = (int)(128u << 16);                    // tensor_dim1 = 128
  g1[3] = (int)(32u << 16);                     // tile_dim0 = 32
  g1[4] = (int)(128u);                          // tile_dim1 = 128, tile_dim2 = 0
  g1[5] = lda;                                  // tensor_dim0_stride (elements)
  g1[6] = 0;
  g1[7] = 0;
  v4i z = {0, 0, 0, 0};
#if __clang_major__ >= 23
  v8i z8 = {0, 0, 0, 0, 0, 0, 0, 0};
  __builtin_amdgcn_tensor_load_to_lds(g0, g1, z, z, z8, 0);
#else
  __builtin_amdgcn_tensor_load_to_lds(g0, g1, z, z, 0);
#endif
#else
  (void)gsrc; (void)ldsAddr; (void)lda;
#endif
}

// ------------------------- shared WMMA mainloop ----------------------------
// 256 threads = 8 waves; block tile 128(M) x 128(N); wave tile 32x64.
// A staged in LDS (TDM, double buffer); B frags direct from packed global.
__device__ inline void gemm_mainloop(const bf16* __restrict__ A, int lda,
                                     const bf16* __restrict__ Bpk, int ktiles,
                                     int m0, int nt0, bf16* sA,
                                     v8f (&acc)[2][4]) {
  int wave = threadIdx.x >> 5;
  int wm = wave & 3, wn = wave >> 2;
#if HAVE_TDM
  if (wave == 0) tdm_load_tileA(A + (size_t)m0 * lda, lds_off(sA), lda);
#endif
  for (int s = 0; s < ktiles; ++s) {
    bf16* cur = sA + (s & 1) * (BLK_M * LDA_S);
#if HAVE_TDM
    bf16* nxt = sA + ((s + 1) & 1) * (BLK_M * LDA_S);
    __syncthreads();                      // nxt buffer free of readers
    if (wave == 0) {
      if (s + 1 < ktiles) {
        tdm_load_tileA(A + (size_t)m0 * lda + (size_t)(s + 1) * BLK_K,
                       lds_off(nxt), lda);
        __builtin_amdgcn_s_wait_tensorcnt(1);   // cur tile complete
      } else {
        __builtin_amdgcn_s_wait_tensorcnt(0);
      }
    }
    __syncthreads();                      // cur visible to all waves
#else
    __syncthreads();
    {   // cooperative fallback copy: 128x32 halves, 8-half chunks
      int t = threadIdx.x;
#pragma unroll
      for (int it = 0; it < 2; ++it) {
        int c = t + it * 256;
        int row = c >> 2, col = (c & 3) * 8;
        *(int4*)(cur + row * LDA_S + col) =
            *(const int4*)(A + (size_t)(m0 + row) * lda + (size_t)s * BLK_K + col);
      }
    }
    __syncthreads();
#endif
    v16bf af[2], bfr[4];
#pragma unroll
    for (int tn = 0; tn < 4; ++tn)
      bfr[tn] = load_b_frag_packed(
          Bpk + ((size_t)(nt0 + wn * 4 + tn) * ktiles + s) * 512);
#pragma unroll
    for (int tm = 0; tm < 2; ++tm)
      af[tm] = load_a_frag(cur + (wm * 32 + tm * 16) * LDA_S);
#pragma unroll
    for (int tm = 0; tm < 2; ++tm)
#pragma unroll
      for (int tn = 0; tn < 4; ++tn)
        acc[tm][tn] = __builtin_amdgcn_wmma_f32_16x16x32_bf16(
            false, af[tm], false, bfr[tn], (short)0, acc[tm][tn], false, false);
  }
}

__device__ inline void zero_acc(v8f (&acc)[2][4]) {
#pragma unroll
  for (int tm = 0; tm < 2; ++tm)
#pragma unroll
    for (int tn = 0; tn < 4; ++tn)
#pragma unroll
      for (int r = 0; r < 8; ++r) acc[tm][tn][r] = 0.0f;
}

// --------------------------------- GEMM1 -----------------------------------
__global__ __launch_bounds__(256) void gemm1_kernel(
    const bf16* __restrict__ A, const bf16* __restrict__ Wpk,
    const float* __restrict__ bias, bf16* __restrict__ Cout) {
  __shared__ __align__(16) bf16 sA[2 * BLK_M * LDA_S];
  int m0 = blockIdx.y * BLK_M, nt0 = blockIdx.x * 8;
  v8f acc[2][4];
  zero_acc(acc);
  gemm_mainloop(A, DIN, Wpk, DIN / 32, m0, nt0, sA, acc);
  int lane = threadIdx.x & 31, wave = threadIdx.x >> 5;
  int wm = wave & 3, wn = wave >> 2;
  int hi = (lane >> 4) * 8, nn = lane & 15;
#pragma unroll
  for (int tm = 0; tm < 2; ++tm)
#pragma unroll
    for (int tn = 0; tn < 4; ++tn)
#pragma unroll
      for (int r = 0; r < 8; ++r) {
        int row = m0 + wm * 32 + tm * 16 + hi + r;
        int col = nt0 * 16 + wn * 64 + tn * 16 + nn;
        float v = acc[tm][tn][r] + bias[col];
        Cout[(size_t)row * Hdim + col] = (bf16)fmaxf(v, 0.0f);
      }
}

// --------------------------------- GEMM2 -----------------------------------
__global__ __launch_bounds__(256) void gemm2_kernel(
    const bf16* __restrict__ A, const bf16* __restrict__ Wpk,
    const float* __restrict__ bias, float* __restrict__ fv) {
  __shared__ __align__(16) bf16 sA[2 * BLK_M * LDA_S];
  int m0 = blockIdx.y * BLK_M;
  v8f acc[2][4];
  zero_acc(acc);
  gemm_mainloop(A, Hdim, Wpk, Hdim / 32, m0, 0, sA, acc);
  int lane = threadIdx.x & 31, wave = threadIdx.x >> 5;
  int wm = wave & 3, wn = wave >> 2;
  int hi = (lane >> 4) * 8, nn = lane & 15;
#pragma unroll
  for (int tm = 0; tm < 2; ++tm)
#pragma unroll
    for (int tn = 0; tn < 4; ++tn)
#pragma unroll
      for (int r = 0; r < 8; ++r) {
        int row = m0 + wm * 32 + tm * 16 + hi + r;
        int col = wn * 64 + tn * 16 + nn;
        fv[(size_t)row * NZ + col] = acc[tm][tn][r] + bias[col];
      }
}

// ------------------------ GEMM3 + distance + exp ---------------------------
__global__ __launch_bounds__(256) void gemm3_kernel(
    const bf16* __restrict__ A, const bf16* __restrict__ Cpk,
    const float* __restrict__ fn, const float* __restrict__ cn,
    float* __restrict__ Mexp) {
  __shared__ __align__(16) bf16 sA[2 * BLK_M * LDA_S];
  int m0 = blockIdx.y * BLK_M, nt0 = blockIdx.x * 8;
  v8f acc[2][4];
  zero_acc(acc);
  gemm_mainloop(A, NZ, Cpk, NZ / 32, m0, nt0, sA, acc);
  int lane = threadIdx.x & 31, wave = threadIdx.x >> 5;
  int wm = wave & 3, wn = wave >> 2;
  int hi = (lane >> 4) * 8, nn = lane & 15;
#pragma unroll
  for (int tm = 0; tm < 2; ++tm)
#pragma unroll
    for (int tn = 0; tn < 4; ++tn)
#pragma unroll
      for (int r = 0; r < 8; ++r) {
        int row = m0 + wm * 32 + tm * 16 + hi + r;
        int col = nt0 * 16 + wn * 64 + tn * 16 + nn;
        float sq = fn[row] + cn[col] - 2.0f * acc[tm][tn][r];
        float d  = sqrtf(fmaxf(sq, 1e-12f));
        Mexp[(size_t)row * NCL + col] = __expf(-d * (1.0f / EPSI));
      }
}

// ------------------------------ small kernels ------------------------------
__global__ void cast_bf16_kernel(const float* s, bf16* d, int n) {
  int i = blockIdx.x * 256 + threadIdx.x;
  if (i < n) d[i] = (bf16)s[i];
}

// Pack B [K,N] (or transposed source [N,K]) into fragment layout:
// dst[(nt*(K/32)+kt)*512 + n*32 + k] = B[kt*32+k][nt*16+n]
__global__ void pack_b_kernel(const float* __restrict__ src, bf16* __restrict__ dst,
                              int K, int N, int srcTransposed) {
  int i = blockIdx.x * 256 + threadIdx.x;
  if (i >= K * N) return;
  int k = i & 31;
  int n = (i >> 5) & 15;
  int blk = i >> 9;
  int ktiles = K >> 5;
  int kt = blk % ktiles, nt = blk / ktiles;
  int gk = kt * 32 + k, gn = nt * 16 + n;
  float v = srcTransposed ? src[(size_t)gn * K + gk] : src[(size_t)gk * N + gn];
  dst[i] = (bf16)v;
}

__global__ void cnorm_kernel(const float* __restrict__ cent, float* __restrict__ cn) {
  int c = blockIdx.x, k = threadIdx.x;          // 1024 blocks x 128 threads
  float v = cent[(size_t)c * NZ + k];
  __shared__ float red[128];
  red[k] = v * v;
  __syncthreads();
  for (int s = 64; s; s >>= 1) { if (k < s) red[k] += red[k + s]; __syncthreads(); }
  if (k == 0) cn[c] = red[0];
}

__global__ void fnorm_kernel(const float* __restrict__ fv,
                             float* __restrict__ fn, bf16* __restrict__ fvbf) {
  int wave = threadIdx.x >> 5, lane = threadIdx.x & 31;
  int row = blockIdx.x * 8 + wave;
  float4 m = ((const float4*)(fv + (size_t)row * NZ))[lane];
  float s = m.x * m.x + m.y * m.y + m.z * m.z + m.w * m.w;
  bf16* q = fvbf + (size_t)row * NZ + lane * 4;
  q[0] = (bf16)m.x; q[1] = (bf16)m.y; q[2] = (bf16)m.z; q[3] = (bf16)m.w;
  for (int off = 16; off; off >>= 1) s += __shfl_down(s, off);
  if (lane == 0) fn[row] = s;
}

__global__ void zero_f_kernel(float* p, int n) {
  int i = blockIdx.x * 256 + threadIdx.x;
  if (i < n) p[i] = 0.0f;
}
__global__ void zero_i_kernel(int* p, int n) {
  int i = blockIdx.x * 256 + threadIdx.x;
  if (i < n) p[i] = 0;
}
__global__ void fill_f_kernel(float* p, float v, int n) {
  int i = blockIdx.x * 256 + threadIdx.x;
  if (i < n) p[i] = v;
}

// v_out[b] = scale / sum_n Mexp[b,n]*u[n]    (row dot; M is L2-resident)
__global__ __launch_bounds__(256) void rowdot_kernel(
    const float* __restrict__ Mexp, const float* __restrict__ u,
    float* __restrict__ vout, float scale) {
  __shared__ float su[NCL];
  for (int i = threadIdx.x; i < NCL; i += 256) su[i] = u[i];
  __syncthreads();
  int wave = threadIdx.x >> 5, lane = threadIdx.x & 31;
  int row = blockIdx.x * 8 + wave;
  const float4* mrow = (const float4*)(Mexp + (size_t)row * NCL);
  float s = 0.0f;
#pragma unroll
  for (int j = 0; j < 8; ++j) {
    int q = lane + 32 * j;
    float4 m = mrow[q];
    int n = q * 4;
    s += m.x * su[n] + m.y * su[n + 1] + m.z * su[n + 2] + m.w * su[n + 3];
  }
  for (int off = 16; off; off >>= 1) s += __shfl_down(s, off);
  if (lane == 0) vout[row] = scale / s;
}

// s1[k] += sum_b Mexp[b,k]*v[b]   (column reduction, 128 rows per block)
__global__ __launch_bounds__(256) void colsum_kernel(
    const float* __restrict__ Mexp, const float* __restrict__ v,
    float* __restrict__ s1) {
  int rb = blockIdx.x * 128, t = threadIdx.x;
  float a0 = 0, a1 = 0, a2 = 0, a3 = 0;
  for (int r = 0; r < 128; ++r) {
    const float* mrow = Mexp + (size_t)(rb + r) * NCL;
    float vb = v[rb + r];
    a0 += mrow[t] * vb;        a1 += mrow[t + 256] * vb;
    a2 += mrow[t + 512] * vb;  a3 += mrow[t + 768] * vb;
  }
  atomicAdd(&s1[t], a0);       atomicAdd(&s1[t + 256], a1);
  atomicAdd(&s1[t + 512], a2); atomicAdd(&s1[t + 768], a3);
}

__global__ void u_update_kernel(const float* s1, float* u) {
  int i = blockIdx.x * 256 + threadIdx.x;
  if (i < NCL) u[i] = (1.0f / NCL) / s1[i];
}

// argmins, soft counts, loss  (dist recovered as -eps*ln(M))
__global__ __launch_bounds__(256) void finalize_kernel(
    const float* __restrict__ Mexp, const float* __restrict__ u,
    const float* __restrict__ v, int* __restrict__ assign,
    float* __restrict__ loss_sum, float* __restrict__ softc,
    int* __restrict__ rawc, int* __restrict__ clusc) {
  __shared__ float su[NCL];
  __shared__ float sc[NCL];
  for (int i = threadIdx.x; i < NCL; i += 256) { su[i] = u[i]; sc[i] = 0.0f; }
  __syncthreads();
  int wave = threadIdx.x >> 5, lane = threadIdx.x & 31;
  int row = blockIdx.x * 8 + wave;
  const float* mrow = Mexp + (size_t)row * NCL;
  float vb = v[row];
  float bs = __int_as_float(0x7f800000); int bsi = 0;   // min soft
  float bm = -__int_as_float(0x7f800000); int bmi = 0;  // max M (= min dist)
#pragma unroll 4
  for (int j = 0; j < NCL / 32; ++j) {
    int n = lane + 32 * j;
    float m = mrow[n];
    float t = m * su[n];
    atomicAdd(&sc[n], t * vb);
    if (t < bs || (t == bs && n < bsi)) { bs = t; bsi = n; }
    if (m > bm || (m == bm && n < bmi)) { bm = m; bmi = n; }
  }
  for (int off = 16; off; off >>= 1) {
    float os = __shfl_down(bs, off); int oi = __shfl_down(bsi, off);
    if (os < bs || (os == bs && oi < bsi)) { bs = os; bsi = oi; }
    float om = __shfl_down(bm, off); int omi = __shfl_down(bmi, off);
    if (om > bm || (om == bm && omi < bmi)) { bm = om; bmi = omi; }
  }
  if (lane == 0) {
    assign[row] = bsi;
    atomicAdd(clusc + bsi, 1);
    atomicAdd(rawc + bmi, 1);
    float m_at = fmaxf(bs / su[bsi], 1e-38f);
    atomicAdd(loss_sum, -EPSI * __logf(m_at));
  }
  __syncthreads();
  for (int i = threadIdx.x; i < NCL; i += 256) atomicAdd(&softc[i], sc[i]);
}

__global__ void outputs_kernel(const float* loss_sum, const int* assign,
                               const float* softc, const int* raw_in,
                               const int* rawc, const int* clus_in,
                               const int* clusc, const float* tsoft_in,
                               float* out) {
  int i = blockIdx.x * 256 + threadIdx.x;
  if (i == 0) out[0] = loss_sum[0] * (1.0f / Bsz);
  if (i < Bsz) out[1 + i] = (float)assign[i];
  if (i < NCL) {
    out[1 + Bsz + i]           = softc[i];
    out[1 + Bsz + NCL + i]     = (float)(raw_in[i] + rawc[i]);
    out[1 + Bsz + 2 * NCL + i] = (float)(clus_in[i] + clusc[i]);
    out[1 + Bsz + 3 * NCL + i] = tsoft_in[i] + softc[i];
  }
}

// --------------------------------- launch ----------------------------------
extern "C" void kernel_launch(void* const* d_in, const int* in_sizes, int n_in,
                              void* d_out, int out_size, void* d_ws,
                              size_t ws_size, hipStream_t stream) {
  const float* x    = (const float*)d_in[0];
  const float* W1   = (const float*)d_in[1];
  const float* b1   = (const float*)d_in[2];
  const float* W2   = (const float*)d_in[3];
  const float* b2   = (const float*)d_in[4];
  const float* cent = (const float*)d_in[5];
  const int*   clin = (const int*)d_in[6];
  const int*   rwin = (const int*)d_in[7];
  const float* tsin = (const float*)d_in[8];
  float* out = (float*)d_out;

  char* ws = (char*)d_ws;
  size_t off = 0;
  auto alloc = [&](size_t bytes) {
    size_t o = off;
    off = (off + bytes + 255) & ~(size_t)255;
    return o;
  };
  bf16*  xbf   = (bf16*)(ws + alloc((size_t)Bsz * DIN * 2));
  bf16*  w1pk  = (bf16*)(ws + alloc((size_t)DIN * Hdim * 2));
  bf16*  w2pk  = (bf16*)(ws + alloc((size_t)Hdim * NZ * 2));
  bf16*  cpk   = (bf16*)(ws + alloc((size_t)NZ * NCL * 2));
  bf16*  Hbf   = (bf16*)(ws + alloc((size_t)Bsz * Hdim * 2));
  float* fv32  = (float*)(ws + alloc((size_t)Bsz * NZ * 4));
  bf16*  fvbf  = (bf16*)(ws + alloc((size_t)Bsz * NZ * 2));
  float* fn    = (float*)(ws + alloc((size_t)Bsz * 4));
  float* cn    = (float*)(ws + alloc((size_t)NCL * 4));
  float* Mexp  = (float*)(ws + alloc((size_t)Bsz * NCL * 4));
  float* u     = (float*)(ws + alloc((size_t)NCL * 4));
  float* s1    = (float*)(ws + alloc((size_t)NCL * 4));
  float* v     = (float*)(ws + alloc((size_t)Bsz * 4));
  int*   assign= (int*)(ws + alloc((size_t)Bsz * 4));
  float* lsum  = (float*)(ws + alloc(256));
  float* softc = (float*)(ws + alloc((size_t)NCL * 4));
  int*   rawc  = (int*)(ws + alloc((size_t)NCL * 4));
  int*   clusc = (int*)(ws + alloc((size_t)NCL * 4));

  // operand preparation
  cast_bf16_kernel<<<(Bsz * DIN + 255) / 256, 256, 0, stream>>>(x, xbf, Bsz * DIN);
  pack_b_kernel<<<(DIN * Hdim + 255) / 256, 256, 0, stream>>>(W1, w1pk, DIN, Hdim, 0);
  pack_b_kernel<<<(Hdim * NZ + 255) / 256, 256, 0, stream>>>(W2, w2pk, Hdim, NZ, 0);
  pack_b_kernel<<<(NZ * NCL + 255) / 256, 256, 0, stream>>>(cent, cpk, NZ, NCL, 1);
  cnorm_kernel<<<NCL, 128, 0, stream>>>(cent, cn);

  // backbone
  gemm1_kernel<<<dim3(Hdim / 128, Bsz / BLK_M), 256, 0, stream>>>(xbf, w1pk, b1, Hbf);
  gemm2_kernel<<<dim3(1, Bsz / BLK_M), 256, 0, stream>>>(Hbf, w2pk, b2, fv32);
  fnorm_kernel<<<Bsz / 8, 256, 0, stream>>>(fv32, fn, fvbf);
  gemm3_kernel<<<dim3(NCL / 128, Bsz / BLK_M), 256, 0, stream>>>(fvbf, cpk, fn, cn, Mexp);

  // sinkhorn: Q = diag(u) M diag(v)
  fill_f_kernel<<<(NCL + 255) / 256, 256, 0, stream>>>(u, 1.0f, NCL);
  rowdot_kernel<<<Bsz / 8, 256, 0, stream>>>(Mexp, u, v, 1.0f);     // init col-norm
  for (int it = 0; it < NIT; ++it) {
    zero_f_kernel<<<(NCL + 255) / 256, 256, 0, stream>>>(s1, NCL);
    colsum_kernel<<<Bsz / 128, 256, 0, stream>>>(Mexp, v, s1);
    u_update_kernel<<<(NCL + 255) / 256, 256, 0, stream>>>(s1, u);
    rowdot_kernel<<<Bsz / 8, 256, 0, stream>>>(Mexp, u, v, 1.0f / Bsz);
  }
  rowdot_kernel<<<Bsz / 8, 256, 0, stream>>>(Mexp, u, v, 1.0f);     // final col-norm

  // reductions / outputs
  zero_f_kernel<<<1, 256, 0, stream>>>(lsum, 1);
  zero_f_kernel<<<(NCL + 255) / 256, 256, 0, stream>>>(softc, NCL);
  zero_i_kernel<<<(NCL + 255) / 256, 256, 0, stream>>>(rawc, NCL);
  zero_i_kernel<<<(NCL + 255) / 256, 256, 0, stream>>>(clusc, NCL);
  finalize_kernel<<<Bsz / 8, 256, 0, stream>>>(Mexp, u, v, assign, lsum, softc, rawc, clusc);
  outputs_kernel<<<(Bsz + 255) / 256, 256, 0, stream>>>(lsum, assign, softc, rwin,
                                                        rawc, clin, clusc, tsin, out);
}
